// YOLO2Boxes_78022375899239
// MI455X (gfx1250) — 8-bit matrix operand order repaired and checked by reading, compiled, not run on hardware
//
#include <hip/hip_runtime.h>

#define NBOX 8192
#define NCLS 80
#define NWRD 256
#define CONF_THRF 0.5f

typedef __attribute__((ext_vector_type(16))) int   v16i;
typedef __attribute__((ext_vector_type(8)))  float v8f;

__device__ __forceinline__ void async_b128(unsigned lds_addr, const void* gaddr) {
  asm volatile("global_load_async_to_lds_b128 %0, %1, off"
               :: "v"(lds_addr), "v"((unsigned long long)(size_t)gaddr)
               : "memory");
}
__device__ __forceinline__ void wait_async0() {
  asm volatile("s_wait_asynccnt 0" ::: "memory");
}

__global__ __launch_bounds__(256) void nms_argmax(const float* __restrict__ logits,
                                                  int* __restrict__ cls) {
  int i = blockIdx.x * 256 + threadIdx.x;
  if (i >= NBOX) return;
  const float4* row = (const float4*)(logits + (size_t)i * NCLS);
  float best = -3.4e38f;
  int bi = 0;
  #pragma unroll 5
  for (int q = 0; q < NCLS / 4; ++q) {
    float4 v = row[q];
    if (v.x > best) { best = v.x; bi = q * 4 + 0; }
    if (v.y > best) { best = v.y; bi = q * 4 + 1; }
    if (v.z > best) { best = v.z; bi = q * 4 + 2; }
    if (v.w > best) { best = v.w; bi = q * 4 + 3; }
  }
  cls[i] = bi;
}

__global__ __launch_bounds__(1024) void nms_sortgather(
    const float* __restrict__ box, const float* __restrict__ conf,
    const int* __restrict__ cls, int* __restrict__ order,
    float* __restrict__ sX1, float* __restrict__ sY1,
    float* __restrict__ sX2, float* __restrict__ sY2,
    int* __restrict__ sCls, int* __restrict__ sVal) {
  __shared__ float skey[NBOX];
  __shared__ int   sidx[NBOX];
  const int tid = threadIdx.x;

  for (int e = tid; e < NBOX; e += 1024) {
    float c = conf[e];
    skey[e] = (c > CONF_THRF) ? -c : 1e30f;
    sidx[e] = e;
  }
  for (unsigned k = 2; k <= NBOX; k <<= 1) {
    for (unsigned j = k >> 1; j; j >>= 1) {
      __syncthreads();
      for (unsigned i = tid; i < NBOX; i += 1024) {
        unsigned p = i ^ j;
        if (p > i) {
          float a = skey[i], b = skey[p];
          bool up = ((i & k) == 0);
          if ((a > b) == up) {
            skey[i] = b; skey[p] = a;
            int t = sidx[i]; sidx[i] = sidx[p]; sidx[p] = t;
          }
        }
      }
    }
  }
  __syncthreads();
  for (int e = tid; e < NBOX; e += 1024) {
    int o = sidx[e];
    float4 b = ((const float4*)box)[o];
    float hw = 0.5f * b.z, hh = 0.5f * b.w;
    sX1[e] = b.x - hw; sY1[e] = b.y - hh;
    sX2[e] = b.x + hw; sY2[e] = b.y + hh;
    sCls[e] = cls[o];
    sVal[e] = (conf[o] > CONF_THRF) ? 1 : 0;
    order[e] = o;
  }
}

__global__ __launch_bounds__(256) void nms_mask(
    const float* __restrict__ sX1, const float* __restrict__ sY1,
    const float* __restrict__ sX2, const float* __restrict__ sY2,
    const int* __restrict__ sCls, unsigned* __restrict__ mask) {
  __shared__ float lx1[2048], ly1[2048], lx2[2048], ly2[2048];
  __shared__ int   lcl[2048];
  const int tid  = threadIdx.x;
  const int lane = tid & 31;
  const int wave = tid >> 5;
  const int half = lane >> 4;
  const int mrow = lane & 15;
  const int i0   = blockIdx.x * 16;

  float rx1[8], ry1[8], rx2[8], ry2[8], rar[8];
  #pragma unroll
  for (int r = 0; r < 8; ++r) {
    int rr = i0 + r + 8 * half;
    rx1[r] = sX1[rr]; ry1[r] = sY1[rr];
    rx2[r] = sX2[rr]; ry2[r] = sY2[rr];
    rar[r] = (rx2[r] - rx1[r]) * (ry2[r] - ry1[r]);
  }

  int ca = sCls[i0 + mrow];
  int k6a = ca & 63;
  int selA = (((ca >> 3) & 1) == half)
               ? (((ca >> 6) << 3) | ((k6a >> 4) << 1) | ((ca >> 2) & 1))
               : -1;
  int valA = (int)(0x38u << ((ca & 3) * 8));
  v16i A;
  #pragma unroll
  for (int v = 0; v < 16; ++v) A[v] = (v == selA) ? valA : 0;

  unsigned ax1 = (unsigned)(size_t)lx1, ay1 = (unsigned)(size_t)ly1;
  unsigned ax2 = (unsigned)(size_t)lx2, ay2 = (unsigned)(size_t)ly2;
  unsigned acl = (unsigned)(size_t)lcl;

  for (int chunk = 0; chunk < 4; ++chunk) {
    const int cbase = chunk * 2048;
    #pragma unroll
    for (int rep = 0; rep < 2; ++rep) {
      int fi = (rep * 256 + tid) * 4;
      async_b128(ax1 + fi * 4, sX1 + cbase + fi);
      async_b128(ay1 + fi * 4, sY1 + cbase + fi);
      async_b128(ax2 + fi * 4, sX2 + cbase + fi);
      async_b128(ay2 + fi * 4, sY2 + cbase + fi);
      async_b128(acl + fi * 4, sCls + cbase + fi);
    }
    wait_async0();
    __syncthreads();

    for (int wi = 0; wi < 8; ++wi) {
      const int lw = wave * 8 + wi;
      const int gw = chunk * 64 + lw;
      unsigned myword = 0u;
      #pragma unroll
      for (int sub = 0; sub < 2; ++sub) {
        const int lc = lw * 32 + sub * 16 + mrow;
        float jx1 = lx1[lc], jy1 = ly1[lc], jx2 = lx2[lc], jy2 = ly2[lc];
        int   cb  = lcl[lc];
        float jar = (jx2 - jx1) * (jy2 - jy1);
        int selB = (((cb >> 3) & 1) == half)
                     ? (((cb >> 6) << 3) | (((cb & 63) >> 4) << 1) | ((cb >> 2) & 1))
                     : -1;
        int valB = (int)(0x38u << ((cb & 3) * 8));
        v16i B;
        #pragma unroll
        for (int v = 0; v < 16; ++v) B[v] = (v == selB) ? valB : 0;

        v8f acc = {0.f, 0.f, 0.f, 0.f, 0.f, 0.f, 0.f, 0.f};
        acc = __builtin_amdgcn_wmma_f32_16x16x128_fp8_fp8(A, B, (short)0, acc,
                                                          false, false);
        const int jglob = cbase + lc;
        #pragma unroll
        for (int r = 0; r < 8; ++r) {
          float iw = fminf(rx2[r], jx2) - fmaxf(rx1[r], jx1);
          float ih = fminf(ry2[r], jy2) - fmaxf(ry1[r], jy1);
          iw = fmaxf(iw, 0.0f); ih = fmaxf(ih, 0.0f);
          float inter = iw * ih;
          float uni   = rar[r] + jar - inter;
          int iglob = i0 + r + 8 * half;
          bool cond = (inter > 0.5f * uni) && (acc[r] > 0.5f) && (jglob > iglob);
          unsigned b = (unsigned)__ballot((int)cond);
          unsigned plo = (b & 0xffffu) << (sub * 16);
          unsigned phi = (b >> 16) << (sub * 16);
          if (lane == r)     myword |= plo;
          if (lane == r + 8) myword |= phi;
        }
      }
      if (lane < 16)
        mask[(size_t)(i0 + lane) * NWRD + gw] = myword;
    }
    __syncthreads();
  }
}

__global__ __launch_bounds__(1024) void nms_scan(const unsigned* __restrict__ mask,
                                                 const int* __restrict__ sVal,
                                                 int* __restrict__ keepS) {
  __shared__ unsigned supp[NWRD];
  __shared__ unsigned diag[32];
  __shared__ unsigned vsW;
  __shared__ unsigned keepW;
  const int tid = threadIdx.x;
  if (tid < NWRD) supp[tid] = 0u;
  __syncthreads();

  for (int c = 0; c < NWRD; ++c) {
    if (tid < 32) {
      int i = c * 32 + tid;
      diag[tid] = mask[(size_t)i * NWRD + c];
      unsigned vb = (unsigned)__ballot(sVal[i] != 0);
      if (tid == 0) vsW = vb;
    }
    __syncthreads();
    if (tid == 0) {
      unsigned w = supp[c], kb = 0u, vs = vsW;
      for (int b = 0; b < 32; ++b) {
        if (((vs >> b) & 1u) && !((w >> b) & 1u)) {
          kb |= 1u << b;
          w |= diag[b];
        }
      }
      supp[c] = w;
      keepW = kb;
    }
    __syncthreads();
    unsigned kb = keepW;
    if (tid < 32) keepS[c * 32 + tid] = (kb >> tid) & 1u;
    unsigned rem = kb;
    while (rem) {
      int b = __ffs(rem) - 1;
      rem &= rem - 1;
      size_t rowoff = (size_t)(c * 32 + b) * NWRD;
      for (int w = c + 1 + tid; w < NWRD; w += 1024)
        supp[w] |= mask[rowoff + w];
    }
    __syncthreads();
  }
}

__global__ __launch_bounds__(256) void nms_finalize(
    const float* __restrict__ box, const float* __restrict__ conf,
    const int* __restrict__ cls, const int* __restrict__ order,
    const int* __restrict__ keepS, float* __restrict__ out) {
  int s = blockIdx.x * 256 + threadIdx.x;
  if (s >= NBOX) return;
  int o = order[s];
  float m = keepS[s] ? 1.0f : 0.0f;
  float4 b = ((const float4*)box)[o];
  out[(size_t)o * 5 + 0] = b.x * m;
  out[(size_t)o * 5 + 1] = b.y * m;
  out[(size_t)o * 5 + 2] = b.z * m;
  out[(size_t)o * 5 + 3] = b.w * m;
  out[(size_t)o * 5 + 4] = conf[o] * m;
  out[NBOX * 5 + o] = (float)cls[o];
  out[NBOX * 6 + o] = m;
}

extern "C" void kernel_launch(void* const* d_in, const int* in_sizes, int n_in,
                              void* d_out, int out_size, void* d_ws, size_t ws_size,
                              hipStream_t stream) {
  const float* box    = (const float*)d_in[0];
  const float* conf   = (const float*)d_in[1];
  const float* logits = (const float*)d_in[2];

  char* ws = (char*)d_ws;
  const size_t K32 = 32 * 1024;
  int*      cls   = (int*)     (ws + 0 * K32);
  int*      order = (int*)     (ws + 1 * K32);
  float*    sX1   = (float*)   (ws + 2 * K32);
  float*    sY1   = (float*)   (ws + 3 * K32);
  float*    sX2   = (float*)   (ws + 4 * K32);
  float*    sY2   = (float*)   (ws + 5 * K32);
  int*      sCls  = (int*)     (ws + 6 * K32);
  int*      sVal  = (int*)     (ws + 7 * K32);
  int*      keepS = (int*)     (ws + 8 * K32);
  unsigned* mask  = (unsigned*)(ws + 16 * K32);

  nms_argmax    <<<NBOX / 256, 256, 0, stream>>>(logits, cls);
  nms_sortgather<<<1, 1024, 0, stream>>>(box, conf, cls, order, sX1, sY1, sX2, sY2, sCls, sVal);
  nms_mask      <<<NBOX / 16, 256, 0, stream>>>(sX1, sY1, sX2, sY2, sCls, mask);
  nms_scan      <<<1, 1024, 0, stream>>>(mask, sVal, keepS);
  nms_finalize  <<<NBOX / 256, 256, 0, stream>>>(box, conf, cls, order, keepS, (float*)d_out);
}
